// GAT2_24318104830716
// MI455X (gfx1250) — matrix operand order repaired and checked by reading, not compiled, not run on hardware
//
#include <hip/hip_runtime.h>
#include <hip/hip_bf16.h>
#include <math.h>

#define L     2048
#define E     256
#define HID   512
#define G4    1024   // 4 * per-direction hidden
#define Hh    256    // per-direction hidden
#define NT    11     // tags
#define START 9
#define STOP  10
#define NEGV  (-10000.0f)

typedef __attribute__((ext_vector_type(16))) _Float16 v16h;
typedef __attribute__((ext_vector_type(8)))  _Float16 v8h;
typedef __attribute__((ext_vector_type(8)))  float    v8f;

__device__ __forceinline__ float sigf(float x) { return 1.0f / (1.0f + expf(-x)); }

// ---------------------------------------------------------------------------
// NT GEMM on WMMA: C[M,N] (f32) = A[M,K](f16, row-major) * Bt[N,K]^T
// + optional bias[N], optional ELU.
// One wave computes a 16(M) x 64(N) macro-tile: 4 accumulators share one
// A-fragment per K-chunk (4x A reuse, 4 v_wmma per chunk).
// A-frag: lane(l16,hs): m=mtile+l16; halves {k=hs*16+0..15} (contiguous)
// B-frag: lane(l16,hs): n=ntile+l16; halves {k=hs*16+0..15} (contiguous, NT form)
// ---------------------------------------------------------------------------
__global__ __launch_bounds__(32)
void gemm_nt_wmma(const _Float16* __restrict__ A, const _Float16* __restrict__ Bt,
                  const float* __restrict__ bias, float* __restrict__ C,
                  int M, int N, int K, int act)
{
  const int lane  = threadIdx.x;
  const int l16   = lane & 15;
  const int hs    = lane >> 4;
  const int ntile = blockIdx.x << 6;   // 64-wide N macro-tile
  const int mtile = blockIdx.y << 4;
  (void)M;

  const _Float16* ap  = A  + (size_t)(mtile + l16) * K + hs * 16;
  const _Float16* bp0 = Bt + (size_t)(ntile +  0 + l16) * K + hs * 16;
  const _Float16* bp1 = Bt + (size_t)(ntile + 16 + l16) * K + hs * 16;
  const _Float16* bp2 = Bt + (size_t)(ntile + 32 + l16) * K + hs * 16;
  const _Float16* bp3 = Bt + (size_t)(ntile + 48 + l16) * K + hs * 16;

  v8f acc0 = {}, acc1 = {}, acc2 = {}, acc3 = {};
  for (int kb = 0; kb < K; kb += 32) {
    v8h a0 = *(const v8h*)(ap + kb);
    v8h a1 = *(const v8h*)(ap + kb + 8);
    v16h af;
#pragma unroll
    for (int i = 0; i < 8; ++i) { af[i] = a0[i]; af[i + 8] = a1[i]; }
    v16h bf0 = *(const v16h*)(bp0 + kb);
    v16h bf1 = *(const v16h*)(bp1 + kb);
    v16h bf2 = *(const v16h*)(bp2 + kb);
    v16h bf3 = *(const v16h*)(bp3 + kb);
    acc0 = __builtin_amdgcn_wmma_f32_16x16x32_f16(false, af, false, bf0, (short)0, acc0, false, false);
    acc1 = __builtin_amdgcn_wmma_f32_16x16x32_f16(false, af, false, bf1, (short)0, acc1, false, false);
    acc2 = __builtin_amdgcn_wmma_f32_16x16x32_f16(false, af, false, bf2, (short)0, acc2, false, false);
    acc3 = __builtin_amdgcn_wmma_f32_16x16x32_f16(false, af, false, bf3, (short)0, acc3, false, false);
  }

  const int m0 = mtile + hs * 8;
#pragma unroll
  for (int sub = 0; sub < 4; ++sub) {
    v8f acc = (sub == 0) ? acc0 : (sub == 1) ? acc1 : (sub == 2) ? acc2 : acc3;
    const int   col = ntile + sub * 16 + l16;
    const float bv  = bias ? bias[col] : 0.0f;
#pragma unroll
    for (int r = 0; r < 8; ++r) {
      float v = acc[r] + bv;
      if (act == 1) v = (v > 0.0f) ? v : expm1f(v);   // ELU(alpha=1)
      C[(size_t)(m0 + r) * N + col] = v;
    }
  }
}

// ---------------------------------------------------------------------------
__global__ void embed_gather_h(const int* __restrict__ sent,
                               const float* __restrict__ emb,
                               _Float16* __restrict__ xh)
{
  int idx = blockIdx.x * blockDim.x + threadIdx.x;
  if (idx >= L * E) return;
  int i = idx / E, e = idx - i * E;
  xh[idx] = (_Float16)emb[(size_t)sent[i] * E + e];
}

__global__ void f32_to_f16(const float* __restrict__ src, _Float16* __restrict__ dst, int n)
{
  int idx = blockIdx.x * blockDim.x + threadIdx.x;
  if (idx < n) dst[idx] = (_Float16)src[idx];
}

// src[R][C] (f32) -> dst[C][R] (f16)
__global__ void transpose_f32_to_f16(const float* __restrict__ src,
                                     _Float16* __restrict__ dst, int R, int C)
{
  int idx = blockIdx.x * blockDim.x + threadIdx.x;
  if (idx >= R * C) return;
  int r = idx / C, c = idx - r * C;
  dst[(size_t)c * R + r] = (_Float16)src[idx];
}

// ---------------------------------------------------------------------------
// BiLSTM scan. blockIdx.x = direction (0 fwd, 1 bwd). 1024 threads: thread j
// computes gate pre-activation z_j = XW[pos][j] + dot(whh[j], h_prev).
// Threads 0..255 then apply i,f,g,o and update c,h.
// ---------------------------------------------------------------------------
__global__ __launch_bounds__(1024)
void bilstm_scan(const float* __restrict__ XWf, const float* __restrict__ XWb,
                 const float* __restrict__ whhf, const float* __restrict__ whhb,
                 const float* __restrict__ hc, float* __restrict__ Hout)
{
  __shared__ __align__(16) float h_prev[Hh];
  __shared__ float zsh[G4];

  const int dir = blockIdx.x;
  const float* XW  = dir ? XWb  : XWf;
  const float* whh = dir ? whhb : whhf;
  const int colofs = dir ? Hh : 0;
  const float* h0 = hc + (dir ? 2 : 0) * Hh;
  const float* c0 = hc + (dir ? 3 : 1) * Hh;

  const int j = threadIdx.x;
  float c = (j < Hh) ? c0[j] : 0.0f;
  if (j < Hh) h_prev[j] = h0[j];
  __syncthreads();

  const float4* wrow = (const float4*)(whh + (size_t)j * Hh);
  for (int step = 0; step < L; ++step) {
    const int pos = dir ? (L - 1 - step) : step;
    float z = XW[(size_t)pos * G4 + j];
    const float4* hp = (const float4*)h_prev;
#pragma unroll 8
    for (int k = 0; k < Hh / 4; ++k) {
      float4 w = wrow[k];
      float4 h = hp[k];
      z += w.x * h.x + w.y * h.y + w.z * h.z + w.w * h.w;
    }
    zsh[j] = z;
    __syncthreads();
    if (j < Hh) {
      float ig = sigf(zsh[j]);
      float fg = sigf(zsh[j + Hh]);
      float gg = tanhf(zsh[j + 2 * Hh]);
      float og = sigf(zsh[j + 3 * Hh]);
      c = fg * c + ig * gg;
      float h = og * tanhf(c);
      h_prev[j] = h;
      Hout[(size_t)pos * HID + colofs + j] = h;
    }
    __syncthreads();
  }
}

// ---------------------------------------------------------------------------
// s1[i] = Wh[i].a[:512]  s2[i] = Wh[i].a[512:]
__global__ void gat_scores(const float* __restrict__ Wh, const float* __restrict__ a,
                           float* __restrict__ s1, float* __restrict__ s2)
{
  int i = blockIdx.x * blockDim.x + threadIdx.x;
  if (i >= L) return;
  const float* row = Wh + (size_t)i * HID;
  float a1 = 0.0f, a2 = 0.0f;
  for (int k = 0; k < HID; ++k) { a1 += row[k] * a[k]; a2 += row[k] * a[HID + k]; }
  s1[i] = a1; s2[i] = a2;
}

// Row-wise softmax of e_ij = lrelu(s1_i + s2_j), written directly as f16 probs.
__global__ __launch_bounds__(256)
void attn_softmax(const float* __restrict__ s1, const float* __restrict__ s2,
                  _Float16* __restrict__ attn)
{
  __shared__ float red[256];
  const int i = blockIdx.x;
  const int tid = threadIdx.x;
  const float si = s1[i];

  float m = -3.4e38f;
  for (int j = tid; j < L; j += 256) {
    float e = si + s2[j]; e = (e > 0.0f) ? e : 0.2f * e;
    m = fmaxf(m, e);
  }
  red[tid] = m; __syncthreads();
  for (int s = 128; s > 0; s >>= 1) {
    if (tid < s) red[tid] = fmaxf(red[tid], red[tid + s]);
    __syncthreads();
  }
  m = red[0]; __syncthreads();

  float sum = 0.0f;
  for (int j = tid; j < L; j += 256) {
    float e = si + s2[j]; e = (e > 0.0f) ? e : 0.2f * e;
    sum += expf(e - m);
  }
  red[tid] = sum; __syncthreads();
  for (int s = 128; s > 0; s >>= 1) {
    if (tid < s) red[tid] += red[tid + s];
    __syncthreads();
  }
  const float inv = 1.0f / red[0];

  for (int j = tid; j < L; j += 256) {
    float e = si + s2[j]; e = (e > 0.0f) ? e : 0.2f * e;
    attn[(size_t)i * L + j] = (_Float16)(expf(e - m) * inv);
  }
}

// feats[i][t] = h2[i].w_tag[t] + b_tag[t]
__global__ void tag_proj(const float* __restrict__ h2, const float* __restrict__ w,
                         const float* __restrict__ b, float* __restrict__ feats)
{
  int idx = blockIdx.x * blockDim.x + threadIdx.x;
  if (idx >= L * NT) return;
  int i = idx / NT, tg = idx - i * NT;
  const float* row = h2 + (size_t)i * HID;
  const float* wr  = w  + (size_t)tg * HID;
  float acc = b[tg];
  for (int k = 0; k < HID; ++k) acc += row[k] * wr[k];
  feats[idx] = acc;
}

// Viterbi decode: single wave32 block; lane = next-tag.
__global__ __launch_bounds__(32)
void viterbi(const float* __restrict__ feats, const float* __restrict__ trans,
             int* __restrict__ bp, float* __restrict__ out)
{
  __shared__ float fv[NT], fv2[NT];
  const int lane = threadIdx.x;
  if (lane < NT) fv[lane] = (lane == START) ? 0.0f : NEGV;
  __syncthreads();

  for (int t = 0; t < L; ++t) {
    if (lane < NT) {
      float best = -3.4e38f; int bi = 0;
      for (int p = 0; p < NT; ++p) {
        float s = fv[p] + trans[lane * NT + p];
        if (s > best) { best = s; bi = p; }
      }
      bp[t * NT + lane] = bi;
      fv2[lane] = best + feats[t * NT + lane];
    }
    __syncthreads();
    if (lane < NT) fv[lane] = fv2[lane];
    __syncthreads();
  }

  if (lane == 0) {
    float best = -3.4e38f; int bt = 0;
    for (int p = 0; p < NT; ++p) {
      float s = fv[p] + trans[STOP * NT + p];
      if (s > best) { best = s; bt = p; }
    }
    out[0] = best;
    int cur = bt;
    for (int t = L - 1; t >= 0; --t) {
      out[1 + t] = (float)cur;
      cur = bp[t * NT + cur];
    }
  }
}

// ---------------------------------------------------------------------------
extern "C" void kernel_launch(void* const* d_in, const int* in_sizes, int n_in,
                              void* d_out, int out_size, void* d_ws, size_t ws_size,
                              hipStream_t stream)
{
  (void)in_sizes; (void)n_in; (void)out_size; (void)ws_size;

  const int*   sent = (const int*)  d_in[0];
  const float* emb  = (const float*)d_in[1];
  const float* w1f  = (const float*)d_in[2];
  const float* u1f  = (const float*)d_in[3];
  const float* b1f  = (const float*)d_in[4];
  const float* w1b  = (const float*)d_in[5];
  const float* u1b  = (const float*)d_in[6];
  const float* b1b  = (const float*)d_in[7];
  const float* hc1  = (const float*)d_in[8];
  const float* g1W  = (const float*)d_in[9];
  const float* g1a  = (const float*)d_in[10];
  const float* g2W  = (const float*)d_in[11];
  const float* g2a  = (const float*)d_in[12];
  const float* w2f  = (const float*)d_in[13];
  const float* u2f  = (const float*)d_in[14];
  const float* b2f  = (const float*)d_in[15];
  const float* w2b  = (const float*)d_in[16];
  const float* u2b  = (const float*)d_in[17];
  const float* b2b  = (const float*)d_in[18];
  const float* hc2  = (const float*)d_in[19];
  const float* wtag = (const float*)d_in[20];
  const float* btag = (const float*)d_in[21];
  const float* trn  = (const float*)d_in[22];
  float* out = (float*)d_out;

  char* ws = (char*)d_ws;
  size_t off = 0;
  auto take = [&](size_t bytes) -> char* {
    char* p = ws + off;
    off += (bytes + 255) & ~(size_t)255;
    return p;
  };
  _Float16* xh    = (_Float16*)take((size_t)L * E * 2);
  _Float16* w1fh  = (_Float16*)take((size_t)G4 * E * 2);
  _Float16* w1bh  = (_Float16*)take((size_t)G4 * E * 2);
  float*    xwf   = (float*)   take((size_t)L * G4 * 4);
  float*    xwb   = (float*)   take((size_t)L * G4 * 4);
  float*    hseq  = (float*)   take((size_t)L * HID * 4);   // h1, later h2
  _Float16* acth  = (_Float16*)take((size_t)L * HID * 2);   // h1_h / g1_h / g2_h
  _Float16* wth   = (_Float16*)take((size_t)HID * HID * 2); // GAT W transposed f16
  float*    whm   = (float*)   take((size_t)L * HID * 4);   // Wh
  _Float16* whth  = (_Float16*)take((size_t)HID * L * 2);   // Wh^T f16
  float*    s1    = (float*)   take((size_t)L * 4);
  float*    s2    = (float*)   take((size_t)L * 4);
  _Float16* attn  = (_Float16*)take((size_t)L * L * 2);
  float*    gout  = (float*)   take((size_t)L * HID * 4);   // g1 then g2
  _Float16* w2fh  = (_Float16*)take((size_t)G4 * HID * 2);
  _Float16* w2bh  = (_Float16*)take((size_t)G4 * HID * 2);
  float*    feats = (float*)   take((size_t)L * NT * 4);
  int*      bp    = (int*)     take((size_t)L * NT * 4);

  dim3 b256(256);

  // --- Embedding + LSTM1 input projections ---
  embed_gather_h<<<(L * E + 255) / 256, b256, 0, stream>>>(sent, emb, xh);
  f32_to_f16<<<(G4 * E + 255) / 256, b256, 0, stream>>>(w1f, w1fh, G4 * E);
  f32_to_f16<<<(G4 * E + 255) / 256, b256, 0, stream>>>(w1b, w1bh, G4 * E);
  gemm_nt_wmma<<<dim3(G4 / 64, L / 16), 32, 0, stream>>>(xh, w1fh, b1f, xwf, L, G4, E, 0);
  gemm_nt_wmma<<<dim3(G4 / 64, L / 16), 32, 0, stream>>>(xh, w1bh, b1b, xwb, L, G4, E, 0);
  bilstm_scan<<<2, 1024, 0, stream>>>(xwf, xwb, u1f, u1b, hc1, hseq);

  // --- Two GAT layers ---
  for (int layer = 0; layer < 2; ++layer) {
    const float* W   = layer ? g2W  : g1W;
    const float* a   = layer ? g2a  : g1a;
    const float* src = layer ? gout : hseq;
    f32_to_f16<<<(L * HID + 255) / 256, b256, 0, stream>>>(src, acth, L * HID);
    transpose_f32_to_f16<<<(HID * HID + 255) / 256, b256, 0, stream>>>(W, wth, HID, HID);
    gemm_nt_wmma<<<dim3(HID / 64, L / 16), 32, 0, stream>>>(acth, wth, nullptr, whm, L, HID, HID, 0);
    gat_scores<<<(L + 255) / 256, b256, 0, stream>>>(whm, a, s1, s2);
    attn_softmax<<<L, 256, 0, stream>>>(s1, s2, attn);
    transpose_f32_to_f16<<<(L * HID + 255) / 256, b256, 0, stream>>>(whm, whth, L, HID);
    gemm_nt_wmma<<<dim3(HID / 64, L / 16), 32, 0, stream>>>(attn, whth, nullptr, gout, L, HID, L, 1);
  }

  // --- LSTM2 ---
  f32_to_f16<<<(L * HID + 255) / 256, b256, 0, stream>>>(gout, acth, L * HID);
  f32_to_f16<<<(G4 * HID + 255) / 256, b256, 0, stream>>>(w2f, w2fh, G4 * HID);
  f32_to_f16<<<(G4 * HID + 255) / 256, b256, 0, stream>>>(w2b, w2bh, G4 * HID);
  gemm_nt_wmma<<<dim3(G4 / 64, L / 16), 32, 0, stream>>>(acth, w2fh, b2f, xwf, L, G4, HID, 0);
  gemm_nt_wmma<<<dim3(G4 / 64, L / 16), 32, 0, stream>>>(acth, w2bh, b2b, xwb, L, G4, HID, 0);
  bilstm_scan<<<2, 1024, 0, stream>>>(xwf, xwb, u2f, u2b, hc2, hseq);

  // --- Tag projection + Viterbi ---
  tag_proj<<<(L * NT + 255) / 256, b256, 0, stream>>>(hseq, wtag, btag, feats);
  viterbi<<<1, 32, 0, stream>>>(feats, trn, bp, out);
}
